// BitLinearAttention_41403484733539
// MI455X (gfx1250) — compile-verified, matrix operand order checked by reading, not run on hardware
//
#include <hip/hip_runtime.h>
#include <hip/hip_fp16.h>
#include <cstdint>

#define DM 1024
#define SEQ 2048
#define BATCH 2
#define NH 16
#define HD 64
#define MROWS (BATCH*SEQ)
#define EPSV 1e-5f

typedef __attribute__((ext_vector_type(16))) _Float16 v16h;
typedef __attribute__((ext_vector_type(8)))  float    v8f;
typedef __attribute__((ext_vector_type(8)))  int      v8i;
typedef int vgi4 __attribute__((vector_size(16)));

#define AS1 __attribute__((address_space(1)))
#define AS3 __attribute__((address_space(3)))

#if __has_builtin(__builtin_amdgcn_global_load_async_to_lds_b128) && \
    __has_builtin(__builtin_amdgcn_s_wait_asynccnt)
#define USE_ASYNC_LDS 1
#else
#define USE_ASYNC_LDS 0
#endif

__device__ __forceinline__ int kb8(int v, int hi)  { return ((v >> 1) << 4) + (hi << 3) + ((v & 1) << 2); }
__device__ __forceinline__ int kb16(int v, int hi) { return ((v >> 2) << 4) + (hi << 3) + ((v & 3) << 1); }

template <int MASK>
__device__ __forceinline__ float dpp_xor(float x) {
  const int xi = __builtin_bit_cast(int, x);
  const int yi = __builtin_amdgcn_update_dpp(xi, xi, 0x160 | MASK, 0xf, 0xf, true);
  return __builtin_bit_cast(float, yi);
}
__device__ __forceinline__ float rowmax16(float x) {
  x = fmaxf(x, dpp_xor<1>(x)); x = fmaxf(x, dpp_xor<2>(x));
  x = fmaxf(x, dpp_xor<4>(x)); x = fmaxf(x, dpp_xor<8>(x));
  return x;
}
__device__ __forceinline__ float rowsum16(float x) {
  x += dpp_xor<1>(x); x += dpp_xor<2>(x);
  x += dpp_xor<4>(x); x += dpp_xor<8>(x);
  return x;
}

__global__ void __launch_bounds__(256) wmean_kernel(
    const float* __restrict__ w0, const float* __restrict__ w1,
    const float* __restrict__ w2, const float* __restrict__ w3,
    float* __restrict__ wmean)
{
  const float* w = (blockIdx.x == 0) ? w0 : (blockIdx.x == 1) ? w1 : (blockIdx.x == 2) ? w2 : w3;
  __shared__ float red[256];
  float s = 0.f;
  for (int i = threadIdx.x; i < DM * DM; i += 256) s += fabsf(w[i]);
  red[threadIdx.x] = s;
  __syncthreads();
  for (int off = 128; off; off >>= 1) {
    if ((int)threadIdx.x < off) red[threadIdx.x] += red[threadIdx.x + off];
    __syncthreads();
  }
  if (threadIdx.x == 0) wmean[blockIdx.x] = fmaxf(red[0] / (float)(DM * DM), EPSV);
}

__global__ void __launch_bounds__(256) wquant_kernel(
    const float* __restrict__ w0, const float* __restrict__ w1,
    const float* __restrict__ w2, const float* __restrict__ w3,
    const float* __restrict__ wmean, int8_t* __restrict__ wq8)
{
  const int m = blockIdx.y;
  const float* w = (m == 0) ? w0 : (m == 1) ? w1 : (m == 2) ? w2 : w3;
  const float inv = 1.0f / wmean[m];
  const int i = (blockIdx.x * 256 + threadIdx.x) * 4;
  const float4 v4 = *(const float4*)(w + i);
  char4 o;
  o.x = (signed char)fminf(fmaxf(rintf(v4.x * inv), -1.f), 1.f);
  o.y = (signed char)fminf(fmaxf(rintf(v4.y * inv), -1.f), 1.f);
  o.z = (signed char)fminf(fmaxf(rintf(v4.z * inv), -1.f), 1.f);
  o.w = (signed char)fminf(fmaxf(rintf(v4.w * inv), -1.f), 1.f);
  *(char4*)(wq8 + (size_t)m * DM * DM + i) = o;
}

__global__ void __launch_bounds__(256) actquant_kernel(
    const float* __restrict__ x, int8_t* __restrict__ q8, float* __restrict__ deq)
{
  __shared__ float red[256];
  const int row = blockIdx.x;
  const float* xr = x + (size_t)row * DM;
  const int t = threadIdx.x;
  const float4 v4 = *(const float4*)(xr + t * 4);
  red[t] = fmaxf(fmaxf(fabsf(v4.x), fabsf(v4.y)), fmaxf(fabsf(v4.z), fabsf(v4.w)));
  __syncthreads();
  for (int off = 128; off; off >>= 1) {
    if (t < off) red[t] = fmaxf(red[t], red[t + off]);
    __syncthreads();
  }
  const float amax = fmaxf(red[0], EPSV);
  const float sc = 127.f / amax;
  char4 o;
  o.x = (signed char)fminf(fmaxf(rintf(v4.x * sc), -128.f), 127.f);
  o.y = (signed char)fminf(fmaxf(rintf(v4.y * sc), -128.f), 127.f);
  o.z = (signed char)fminf(fmaxf(rintf(v4.z * sc), -128.f), 127.f);
  o.w = (signed char)fminf(fmaxf(rintf(v4.w * sc), -128.f), 127.f);
  *(char4*)(q8 + (size_t)row * DM + t * 4) = o;
  if (t == 0) deq[row] = amax * (1.f / 127.f);
}

template <typename OutT>
__global__ void __launch_bounds__(128) bitproj_kernel(
    const int8_t* __restrict__ aq8, const float* __restrict__ adeq,
    const int8_t* __restrict__ wq8all, const float* __restrict__ wmean,
    int wbase, OutT* __restrict__ outbase)
{
  const int which = wbase + blockIdx.z;
  const int8_t* w8 = wq8all + (size_t)which * DM * DM;
  OutT* out = outbase + (size_t)blockIdx.z * MROWS * DM;
  const float wd = wmean[which];
  const int wave = threadIdx.x >> 5;
  const int lane = threadIdx.x & 31;
  const int m0 = blockIdx.x * 64 + wave * 16;
  const int n0 = blockIdx.y * 64;
  const int r = lane & 15, hi = lane >> 4;

  v8i acc[4] = {};
  const int8_t* xrow = aq8 + (size_t)(m0 + r) * DM;
  for (int k0 = 0; k0 < DM; k0 += 64) {
    v8i a;
#pragma unroll
    for (int vv = 0; vv < 8; ++vv) a[vv] = *(const int*)(xrow + k0 + kb8(vv, hi));
#pragma unroll
    for (int nt = 0; nt < 4; ++nt) {
      const int8_t* wrow = w8 + (size_t)(n0 + nt * 16 + r) * DM;
      v8i bf;
#pragma unroll
      for (int vv = 0; vv < 8; ++vv) bf[vv] = *(const int*)(wrow + k0 + kb8(vv, hi));
      acc[nt] = __builtin_amdgcn_wmma_i32_16x16x64_iu8(true, a, true, bf, acc[nt], false, false);
    }
  }
#pragma unroll
  for (int i = 0; i < 8; ++i) {
    const int mr = m0 + i + 8 * hi;
    const float xd = adeq[mr] * wd;
#pragma unroll
    for (int nt = 0; nt < 4; ++nt)
      out[(size_t)mr * DM + n0 + nt * 16 + r] = (OutT)((float)acc[nt][i] * xd);
  }
}

__global__ void __launch_bounds__(128) attn_kernel(
    const _Float16* __restrict__ q, const _Float16* __restrict__ k,
    const _Float16* __restrict__ v, float* __restrict__ o)
{
  __shared__ _Float16 pShAll[4 * 16 * 32];
  __shared__ _Float16 vShAll[4 * 32 * HD];

  const int wave = threadIdx.x >> 5;
  const int lane = threadIdx.x & 31;
  _Float16* pSh = pShAll + wave * 16 * 32;
  _Float16* vSh = vShAll + wave * 32 * HD;

  const int qt = blockIdx.x * 4 + wave;
  const int h  = blockIdx.y;
  const int b  = blockIdx.z;
  const int r = lane & 15, hi = lane >> 4;

  const size_t rowBase = (size_t)b * SEQ;
  const size_t hOff = (size_t)h * HD;

  v16h qa[2];
#pragma unroll
  for (int c = 0; c < 2; ++c) {
    const _Float16* qp = q + (rowBase + qt * 16 + r) * DM + hOff + c * 32;
    v8i t;
#pragma unroll
    for (int vv = 0; vv < 8; ++vv) t[vv] = *(const int*)(qp + kb16(vv, hi));
    qa[c] = __builtin_bit_cast(v16h, t);
  }

  float mrow[8], lrow[8];
  v8f oAcc[4] = {};
#pragma unroll
  for (int i = 0; i < 8; ++i) { mrow[i] = -1e30f; lrow[i] = 0.f; }

  const int nchunks = (qt * 16 + 16 + 31) / 32;
  for (int ch = 0; ch < nchunks; ++ch) {
    const int j0 = ch * 32;

    const _Float16* vp = v + (rowBase + j0 + lane) * DM + hOff;
#if USE_ASYNC_LDS
    {
      AS1 vgi4* gp = (AS1 vgi4*)vp;
      AS3 vgi4* lp = (AS3 vgi4*)(&vSh[lane * HD]);
#pragma unroll
      for (int t = 0; t < 8; ++t)
        __builtin_amdgcn_global_load_async_to_lds_b128(gp + t, lp + t, 0, 0);
    }
#else
#pragma unroll
    for (int t = 0; t < 8; ++t)
      *(int4*)(&vSh[lane * HD + t * 8]) = *(const int4*)(vp + t * 8);
#endif

    if (ch + 1 < nchunks) {
      __builtin_prefetch(vp + 32 * DM, 0, 1);
      __builtin_prefetch(k + (rowBase + j0 + 32 + lane) * DM + hOff, 0, 1);
    }

    v8f sAcc[2] = {};
#pragma unroll
    for (int nt = 0; nt < 2; ++nt) {
#pragma unroll
      for (int c = 0; c < 2; ++c) {
        const _Float16* kp = k + (rowBase + j0 + nt * 16 + r) * DM + hOff + c * 32;
        v8i t;
#pragma unroll
        for (int vv = 0; vv < 8; ++vv) t[vv] = *(const int*)(kp + kb16(vv, hi));
        v16h kf = __builtin_bit_cast(v16h, t);
        sAcc[nt] = __builtin_amdgcn_wmma_f32_16x16x32_f16(
            false, qa[c], false, kf, (short)0, sAcc[nt], false, false);
      }
    }

#pragma unroll
    for (int nt = 0; nt < 2; ++nt)
#pragma unroll
      for (int i = 0; i < 8; ++i) {
        const int col = j0 + nt * 16 + r;
        const int row = qt * 16 + i + 8 * hi;
        const float s = sAcc[nt][i] * 0.125f;
        sAcc[nt][i] = (col > row) ? -1e30f : s;
      }

    float corr[8];
#pragma unroll
    for (int i = 0; i < 8; ++i) {
      const float mx = rowmax16(fmaxf(sAcc[0][i], sAcc[1][i]));
      const float mnew = fmaxf(mrow[i], mx);
      corr[i] = __expf(mrow[i] - mnew);
      sAcc[0][i] = __expf(sAcc[0][i] - mnew);
      sAcc[1][i] = __expf(sAcc[1][i] - mnew);
      const float sm = rowsum16(sAcc[0][i] + sAcc[1][i]);
      lrow[i] = lrow[i] * corr[i] + sm;
      mrow[i] = mnew;
    }
#pragma unroll
    for (int dt = 0; dt < 4; ++dt)
#pragma unroll
      for (int i = 0; i < 8; ++i) oAcc[dt][i] *= corr[i];

#pragma unroll
    for (int nt = 0; nt < 2; ++nt)
#pragma unroll
      for (int i = 0; i < 8; ++i)
        pSh[(i + 8 * hi) * 32 + nt * 16 + r] = (_Float16)sAcc[nt][i];

    asm volatile("s_wait_dscnt 0x0" ::: "memory");
#if USE_ASYNC_LDS
    __builtin_amdgcn_s_wait_asynccnt(0);
#endif

    v16h pa;
    {
      v8i t;
#pragma unroll
      for (int vv = 0; vv < 8; ++vv) t[vv] = *(const int*)(&pSh[r * 32 + kb16(vv, hi)]);
      pa = __builtin_bit_cast(v16h, t);
    }

#pragma unroll
    for (int dt = 0; dt < 4; ++dt) {
      v8i t;
#pragma unroll
      for (int vv = 0; vv < 8; ++vv) {
        const int kb = kb16(vv, hi);
        const unsigned lo  = *(const unsigned short*)(&vSh[(kb + 0) * HD + dt * 16 + r]);
        const unsigned hi2 = *(const unsigned short*)(&vSh[(kb + 1) * HD + dt * 16 + r]);
        t[vv] = (int)(lo | (hi2 << 16));
      }
      v16h vb = __builtin_bit_cast(v16h, t);
      oAcc[dt] = __builtin_amdgcn_wmma_f32_16x16x32_f16(
          false, pa, false, vb, (short)0, oAcc[dt], false, false);
    }
  }

#pragma unroll
  for (int dt = 0; dt < 4; ++dt)
#pragma unroll
    for (int i = 0; i < 8; ++i) {
      const int row = qt * 16 + i + 8 * hi;
      o[(rowBase + row) * DM + hOff + dt * 16 + r] = oAcc[dt][i] / lrow[i];
    }
}

extern "C" void kernel_launch(void* const* d_in, const int* in_sizes, int n_in,
                              void* d_out, int out_size, void* d_ws, size_t ws_size,
                              hipStream_t stream)
{
  const float* x  = (const float*)d_in[0];
  const float* Wq = (const float*)d_in[2];
  const float* Wk = (const float*)d_in[3];
  const float* Wv = (const float*)d_in[4];
  const float* Wo = (const float*)d_in[5];

  char* ws = (char*)d_ws;
  size_t off = 0;
  int8_t*   xq8   = (int8_t*)(ws + off);   off += (size_t)MROWS * DM;
  float*    xdeq  = (float*)(ws + off);    off += (size_t)MROWS * 4;
  float*    wmean = (float*)(ws + off);    off += 64;
  int8_t*   wq8   = (int8_t*)(ws + off);   off += (size_t)4 * DM * DM;
  _Float16* qkv   = (_Float16*)(ws + off); off += (size_t)3 * MROWS * DM * 2;
  float*    attno = (float*)(ws + off);    off += (size_t)MROWS * DM * 4;
  int8_t*   aq8   = (int8_t*)(ws + off);   off += (size_t)MROWS * DM;
  float*    adeq  = (float*)(ws + off);    off += (size_t)MROWS * 4;

  wmean_kernel<<<4, 256, 0, stream>>>(Wq, Wk, Wv, Wo, wmean);
  wquant_kernel<<<dim3(DM * DM / 1024, 4), 256, 0, stream>>>(Wq, Wk, Wv, Wo, wmean, wq8);
  actquant_kernel<<<MROWS, 256, 0, stream>>>(x, xq8, xdeq);
  bitproj_kernel<_Float16><<<dim3(MROWS / 64, DM / 64, 3), 128, 0, stream>>>(
      xq8, xdeq, wq8, wmean, 0, qkv);
  attn_kernel<<<dim3(SEQ / 64, NH, BATCH), 128, 0, stream>>>(
      qkv, qkv + (size_t)MROWS * DM, qkv + (size_t)2 * MROWS * DM, attno);
  actquant_kernel<<<MROWS, 256, 0, stream>>>(attno, aq8, adeq);
  bitproj_kernel<float><<<dim3(MROWS / 64, DM / 64, 1), 128, 0, stream>>>(
      aq8, adeq, wq8, wmean, 3, (float*)d_out);
}
